// EQSS2D_78065325572544
// MI455X (gfx1250) — compile-verified, matrix operand order checked by reading, not run on hardware
//
#include <hip/hip_runtime.h>
#include <math.h>

// ---- problem constants ----
#define Bb 4
#define Kk 4
#define Dd 192
#define Hh 64
#define Ww 64
#define Ll 4096
#define Nn 16
#define Rr 6
#define RN 38        // R + 2N
#define NCH 64       // number of scan chunks
#define CHT 64       // chunk length (NCH*CHT == Ll)

typedef __attribute__((ext_vector_type(16))) _Float16 v16h;
typedef __attribute__((ext_vector_type(8)))  _Float16 v8h;
typedef __attribute__((ext_vector_type(8)))  float    v8f;

// direction maps: xs[b,k,d,l] = x[b,k,d, src_idx(l,k)]
__device__ __forceinline__ int src_idx(int l, int k) {
    int i = l >> 6, j = l & 63;
    switch (k & 3) {
        case 0:  return l;
        case 1:  return ((63 - j) << 6) + i;        // rot90(k=-1) then flatten
        case 2:  return 4095 - l;                   // reversed
        default: return (j << 6) + (63 - i);        // rot90(k=+1) then flatten
    }
}
// inverse map for cross_merge: ym[b,d,p] += ys[b,k,d, inv_idx(p,k)]
__device__ __forceinline__ int inv_idx(int p, int k) {
    int row = p >> 6, col = p & 63;
    switch (k & 3) {
        case 0:  return p;
        case 1:  return (col << 6) + (63 - row);
        case 2:  return 4095 - p;
        default: return ((63 - col) << 6) + row;
    }
}

__device__ __forceinline__ float softplusf(float z) {
    return (z > 20.0f) ? z : log1pf(__expf(z));
}

// ---------------------------------------------------------------------------
// Kernel 1: pack x_proj_w (K, 38, 192) f32 into WMMA B-fragment layout, f16.
// Fragment element e of lane t (n = t&15, half = t>>4):
//   Klocal = (e<8 ? 0 : 16) + half*8 + (e&7)
// Buffer: [k][kchunk(6)][ntile(3)][lane(32)][16 halfs]; rows r>=38 zero-padded.
// ---------------------------------------------------------------------------
__global__ void pack_w_kernel(const float* __restrict__ xpw, _Float16* __restrict__ wfrag) {
    int tid = blockIdx.x * blockDim.x + threadIdx.x;   // 0..2303
    if (tid >= 4 * 6 * 3 * 32) return;
    int lane  = tid & 31;
    int ntile = (tid >> 5) % 3;
    int kc    = (tid / 96) % 6;
    int kdir  = tid / 576;
    int n    = lane & 15;
    int half = lane >> 4;
    int r = ntile * 16 + n;
    v16h out;
    #pragma unroll
    for (int e = 0; e < 16; ++e) {
        int Kl = ((e < 8) ? 0 : 16) + half * 8 + (e & 7);
        int d  = kc * 32 + Kl;
        float v = (r < RN) ? xpw[(kdir * RN + r) * Dd + d] : 0.0f;
        out[e] = (_Float16)v;
    }
    *(v16h*)(wfrag + (size_t)tid * 16) = out;
}

// ---------------------------------------------------------------------------
// Kernel 2: build xs_t (B,K,L,D) f16 — cross_scan gather + transpose + cvt.
// ---------------------------------------------------------------------------
__global__ void pack_xs_kernel(const float* __restrict__ x, _Float16* __restrict__ xs_t) {
    size_t idx = (size_t)blockIdx.x * blockDim.x + threadIdx.x;
    if (idx >= (size_t)Bb * Kk * Ll * Dd) return;
    int d  = (int)(idx % Dd);
    int l  = (int)((idx / Dd) % Ll);
    int bk = (int)(idx / ((size_t)Dd * Ll));
    int s  = src_idx(l, bk & 3);
    float v = x[((size_t)bk * Dd + d) * Ll + s];
    xs_t[idx] = (_Float16)v;
}

// ---------------------------------------------------------------------------
// Kernel 3: x_dbl_t (B,K,L,38) f32 = (W @ xs)^T per (b,k), via WMMA f16->f32.
// One wave per 16(l) x 16(r) output tile; K-loop of 6 chunks of 32.
// ---------------------------------------------------------------------------
__global__ void gemm_xdbl_kernel(const _Float16* __restrict__ xs_t,
                                 const _Float16* __restrict__ wfrag,
                                 float* __restrict__ xdbl) {
    int wid  = (int)((blockIdx.x * blockDim.x + threadIdx.x) >> 5);
    int lane = threadIdx.x & 31;
    int ltile =  wid & 255;
    int ntile = (wid >> 8) % 3;
    int bk    =  wid / 768;              // b*4 + k
    int kdir  =  bk & 3;
    int n    = lane & 15;
    int half = lane >> 4;

    const _Float16* abase = xs_t + ((size_t)bk * Ll + ltile * 16 + n) * Dd + half * 8;
    const _Float16* bbase = wfrag + (((size_t)(kdir * 6) * 3 + ntile) * 32 + lane) * 16;

    v8f c = {};
    #pragma unroll
    for (int kc = 0; kc < 6; ++kc) {
        v8h lo = *(const v8h*)(abase + kc * 32);
        v8h hi = *(const v8h*)(abase + kc * 32 + 16);
        v16h a = __builtin_shufflevector(lo, hi, 0,1,2,3,4,5,6,7,8,9,10,11,12,13,14,15);
        v16h bf = *(const v16h*)(bbase + (size_t)kc * 3 * 512);
        c = __builtin_amdgcn_wmma_f32_16x16x32_f16(
                /*neg_a=*/false, a, /*neg_b=*/false, bf,
                /*c_mod=*/(short)0, c, /*reuse_a=*/false, /*reuse_b=*/false);
    }

    int r = ntile * 16 + n;
    if (r < RN) {
        #pragma unroll
        for (int j = 0; j < 8; ++j) {
            int l = ltile * 16 + half * 8 + j;      // D-tile row M = j + half*8
            xdbl[((size_t)bk * Ll + l) * RN + r] = c[j];
        }
    }
}

// ---------------------------------------------------------------------------
// Chunked parallel scan of the linear recurrence h_l = a_l*h_{l-1} + b_l.
// Wave layout (phases 1 & 3): lane = state n (16 per channel), 2 channels per
// wave; one wave per (channel-pair, chunk) -> 98304 waves.
// Chunk-state layout: buf[(channel*NCH + chunk)*Nn + n]  (n contiguous).
// ---------------------------------------------------------------------------

// Phase 1: per chunk compute P = prod(a) and S = chunk-local state (zero init).
__global__ void scan_phase1_kernel(const float* __restrict__ x,
                                   const float* __restrict__ xdbl,
                                   const float* __restrict__ dtw,
                                   const float* __restrict__ dtb,
                                   const float* __restrict__ A_log,
                                   float* __restrict__ Pbuf,
                                   float* __restrict__ Sbuf) {
    int gw   = (int)((blockIdx.x * blockDim.x + threadIdx.x) >> 5);
    int lane = threadIdx.x & 31;
    int chunk = gw & (NCH - 1);
    int chp   = gw >> 6;                 // channel pair 0..1535
    int bk = chp / 96;
    int dp = chp % 96;
    int half = lane >> 4, n = lane & 15;
    int d = dp * 2 + half;
    int k = bk & 3;
    int kd = k * Dd + d;

    float An   = -__expf(A_log[kd * Nn + n]);
    float w0 = dtw[kd * Rr + 0], w1 = dtw[kd * Rr + 1], w2 = dtw[kd * Rr + 2];
    float w3 = dtw[kd * Rr + 3], w4 = dtw[kd * Rr + 4], w5 = dtw[kd * Rr + 5];
    float bias = dtb[kd];

    const float* xrow = x    + ((size_t)bk * Dd + d) * Ll;
    const float* xd   = xdbl + (size_t)bk * Ll * RN;

    float P = 1.0f, S = 0.0f;
    int l0 = chunk * CHT;
    for (int t = 0; t < CHT; ++t) {
        int l = l0 + t;
        const float* row = xd + (size_t)l * RN;
        float z = bias + w0 * row[0] + w1 * row[1] + w2 * row[2]
                       + w3 * row[3] + w4 * row[4] + w5 * row[5];
        float dtv = softplusf(z);
        float u   = xrow[src_idx(l, k)];
        float a   = __expf(dtv * An);
        P *= a;
        S = S * a + dtv * u * row[Rr + n];
    }
    int ch = bk * Dd + d;
    size_t base = ((size_t)ch * NCH + chunk) * Nn + n;
    Pbuf[base] = P;
    Sbuf[base] = S;
}

// Phase 2: inter-chunk prefix, in place: Sbuf[c] becomes H_init for chunk c.
__global__ void scan_phase2_kernel(const float* __restrict__ Pbuf,
                                   float* __restrict__ SHbuf) {
    int tid = blockIdx.x * blockDim.x + threadIdx.x;   // 49152 = 3072*16
    if (tid >= Bb * Kk * Dd * Nn) return;
    int n  = tid & (Nn - 1);
    int ch = tid >> 4;
    float h = 0.0f;
    for (int c = 0; c < NCH; ++c) {
        size_t idx = ((size_t)ch * NCH + c) * Nn + n;
        float s = SHbuf[idx];
        float p = Pbuf[idx];
        SHbuf[idx] = h;              // H_init for chunk c
        h = s + p * h;
    }
}

// Phase 3: replay each chunk from its H_init, emit y (16-lane shfl reduction).
__global__ void scan_phase3_kernel(const float* __restrict__ x,
                                   const float* __restrict__ xdbl,
                                   const float* __restrict__ dtw,
                                   const float* __restrict__ dtb,
                                   const float* __restrict__ A_log,
                                   const float* __restrict__ Ds,
                                   const float* __restrict__ Hbuf,
                                   float* __restrict__ ys) {
    int gw   = (int)((blockIdx.x * blockDim.x + threadIdx.x) >> 5);
    int lane = threadIdx.x & 31;
    int chunk = gw & (NCH - 1);
    int chp   = gw >> 6;
    int bk = chp / 96;
    int dp = chp % 96;
    int half = lane >> 4, n = lane & 15;
    int d = dp * 2 + half;
    int k = bk & 3;
    int kd = k * Dd + d;

    float An   = -__expf(A_log[kd * Nn + n]);
    float w0 = dtw[kd * Rr + 0], w1 = dtw[kd * Rr + 1], w2 = dtw[kd * Rr + 2];
    float w3 = dtw[kd * Rr + 3], w4 = dtw[kd * Rr + 4], w5 = dtw[kd * Rr + 5];
    float bias = dtb[kd];
    float dsv  = Ds[kd];

    const float* xrow = x    + ((size_t)bk * Dd + d) * Ll;
    const float* xd   = xdbl + (size_t)bk * Ll * RN;
    float*       yrow = ys   + ((size_t)bk * Dd + d) * Ll;

    int ch = bk * Dd + d;
    float h = Hbuf[((size_t)ch * NCH + chunk) * Nn + n];

    int l0 = chunk * CHT;
    for (int t = 0; t < CHT; ++t) {
        int l = l0 + t;
        const float* row = xd + (size_t)l * RN;
        float z = bias + w0 * row[0] + w1 * row[1] + w2 * row[2]
                       + w3 * row[3] + w4 * row[4] + w5 * row[5];
        float dtv = softplusf(z);
        float u   = xrow[src_idx(l, k)];
        float a   = __expf(dtv * An);
        h = h * a + dtv * u * row[Rr + n];
        float part = h * row[Rr + Nn + n];
        part += __shfl_xor(part, 1, 32);
        part += __shfl_xor(part, 2, 32);
        part += __shfl_xor(part, 4, 32);
        part += __shfl_xor(part, 8, 32);        // sum over 16 states
        if (n == 0) yrow[l] = part + dsv * u;
    }
}

// ---------------------------------------------------------------------------
// Kernel 5: cross_merge (gather-sum over k) + LayerNorm over D.
// ---------------------------------------------------------------------------
__global__ void merge_ln_kernel(const float* __restrict__ ys,
                                const float* __restrict__ lnw,
                                const float* __restrict__ lnb,
                                float* __restrict__ out) {
    int b = blockIdx.x >> 12;
    int p = blockIdx.x & 4095;
    int d = threadIdx.x;

    float acc = 0.0f;
    #pragma unroll
    for (int k = 0; k < 4; ++k) {
        int l = inv_idx(p, k);
        acc += ys[(((size_t)(b * 4 + k)) * Dd + d) * Ll + l];
    }

    __shared__ float s1[Dd];
    __shared__ float s2[Dd];
    s1[d] = acc;
    s2[d] = acc * acc;
    __syncthreads();
    if (d < 64) { s1[d] += s1[d + 64] + s1[d + 128]; s2[d] += s2[d + 64] + s2[d + 128]; }
    __syncthreads();
    for (int s = 32; s > 0; s >>= 1) {
        if (d < s) { s1[d] += s1[d + s]; s2[d] += s2[d + s]; }
        __syncthreads();
    }
    float mu  = s1[0] * (1.0f / Dd);
    float var = s2[0] * (1.0f / Dd) - mu * mu;
    float rs  = rsqrtf(var + 1e-5f);
    out[(size_t)blockIdx.x * Dd + d] = (acc - mu) * rs * lnw[d] + lnb[d];
}

// ---------------------------------------------------------------------------
extern "C" void kernel_launch(void* const* d_in, const int* in_sizes, int n_in,
                              void* d_out, int out_size, void* d_ws, size_t ws_size,
                              hipStream_t stream) {
    const float* x     = (const float*)d_in[0];
    const float* xpw   = (const float*)d_in[1];
    const float* dtw   = (const float*)d_in[2];
    const float* dtb   = (const float*)d_in[3];
    const float* A_log = (const float*)d_in[4];
    const float* Ds    = (const float*)d_in[5];
    const float* lnw   = (const float*)d_in[6];
    const float* lnb   = (const float*)d_in[7];
    float* out = (float*)d_out;

    char* ws = (char*)d_ws;
    const size_t OFF_WFRAG = 0;                       //     73,728 B (f16 B-fragments)
    const size_t OFF_XST   = 131072;                  // 25,165,824 B (xs_t f16)
    const size_t OFF_XDBL  = OFF_XST  + 25165824;     //  9,961,472 B (x_dbl_t f32)
    const size_t OFF_YS    = OFF_XDBL + 9961472;      // 50,331,648 B (ys f32)
    const size_t OFF_P     = OFF_YS   + 50331648;     // 12,582,912 B (chunk products)
    const size_t OFF_SH    = OFF_P    + 12582912;     // 12,582,912 B (chunk states / H)
    _Float16* wfrag = (_Float16*)(ws + OFF_WFRAG);
    _Float16* xs_t  = (_Float16*)(ws + OFF_XST);
    float*    xdbl  = (float*)(ws + OFF_XDBL);
    float*    ysb   = (float*)(ws + OFF_YS);
    float*    Pbuf  = (float*)(ws + OFF_P);
    float*    SHbuf = (float*)(ws + OFF_SH);

    pack_w_kernel     <<<9, 256, 0, stream>>>(xpw, wfrag);
    pack_xs_kernel    <<<(Bb * Kk * Ll * Dd) / 256, 256, 0, stream>>>(x, xs_t);
    gemm_xdbl_kernel  <<<1536, 256, 0, stream>>>(xs_t, wfrag, xdbl);
    // chunked parallel scan: 98304 waves for phases 1/3
    scan_phase1_kernel<<<12288, 256, 0, stream>>>(x, xdbl, dtw, dtb, A_log, Pbuf, SHbuf);
    scan_phase2_kernel<<<192, 256, 0, stream>>>(Pbuf, SHbuf);
    scan_phase3_kernel<<<12288, 256, 0, stream>>>(x, xdbl, dtw, dtb, A_log, Ds, SHbuf, ysb);
    merge_ln_kernel   <<<Bb * Ll, Dd, 0, stream>>>(ysb, lnw, lnb, out);
}
